// SoftGroupAttention_76149770158125
// MI455X (gfx1250) — matrix operand order repaired and checked by reading, not compiled, not run on hardware
//
#include <hip/hip_runtime.h>
#include <hip/hip_bf16.h>
#include <math.h>

// ---------------------------------------------------------------------------
// SoftGroupAttention for MI455X (gfx1250, wave32, WMMA f32_16x16x32_f16)
// Async global->LDS staged, double-buffered B tiles; A-fragment prefetch.
// ---------------------------------------------------------------------------
typedef __attribute__((ext_vector_type(16))) _Float16 v16h;
typedef __attribute__((ext_vector_type(8)))  _Float16 v8h;
typedef __attribute__((ext_vector_type(8)))  float    v8f;
typedef __attribute__((ext_vector_type(4)))  int      v4i_t;

#define B_SZ   8
#define N_SEQ  2048
#define DIM    768
#define TD     2304          // 3*DIM
#define MROWS  16384         // B_SZ * N_SEQ
#define GNUM   49
#define GPAD   64
#define SLOT   24            // halves per LDS fragment slot (48B: 16B-aligned, 2-way banks)

#if defined(__has_builtin)
#if __has_builtin(__builtin_amdgcn_global_load_async_to_lds_b128)
#define HAVE_ASYNC_LDS 1
#endif
#endif
#ifndef HAVE_ASYNC_LDS
#define HAVE_ASYNC_LDS 0
#endif

#if HAVE_ASYNC_LDS
typedef __attribute__((address_space(1))) v4i_t gvec4i;
typedef __attribute__((address_space(3))) v4i_t lvec4i;

// async copy of 16 bytes (8 halves) global -> LDS
__device__ __forceinline__ void async_copy16(const _Float16* g, _Float16* l) {
  v4i_t* gg = const_cast<v4i_t*>((const v4i_t*)g);  // reinterpret (generic)
  v4i_t* ll = (v4i_t*)l;
  __builtin_amdgcn_global_load_async_to_lds_b128((gvec4i*)gg, (lvec4i*)ll, 0, 0);
}
#endif

__device__ __forceinline__ void async_fence() {
#if HAVE_ASYNC_LDS
  asm volatile("s_wait_asynccnt 0" ::: "memory");
#endif
}

__device__ __forceinline__ v8f wmma32(const v16h& a, const v16h& b, const v8f& c) {
  return __builtin_amdgcn_wmma_f32_16x16x32_f16(false, a, false, b, (short)0, c,
                                                false, false);
}

// A fragment (16x32 f16): lane l<16 -> row=row0+l, K chunks {k..k+7},{k+16..k+23}
__device__ __forceinline__ v16h load_A(const _Float16* base, int lda, int row0, int k0) {
  int lane = threadIdx.x & 31;
  const _Float16* p = base + (size_t)(row0 + (lane & 15)) * lda + k0 + ((lane >> 4) * 8);
  v8h lo = *(const v8h*)p;
  v8h hi = *(const v8h*)(p + 16);
  v16h r;
#pragma unroll
  for (int i = 0; i < 8; ++i) { r[i] = lo[i]; r[i + 8] = hi[i]; }
  return r;
}

// Cooperative stage of a (JT*16 cols x 32 K) B tile into LDS in fragment layout:
// slot (j,lane) = 16 halves at sbuf + (j*32+lane)*SLOT (48B stride).
template <int JT>
__device__ __forceinline__ void stage_B(const _Float16* Bt, int ldb, int colb, int k0,
                                        _Float16* sbuf) {
  int t = threadIdx.x;
  if (t < JT * 32) {
    int j = t >> 5, lane = t & 31;
    const _Float16* p =
        Bt + (size_t)(colb + j * 16 + (lane & 15)) * ldb + k0 + ((lane >> 4) * 8);
    _Float16* l = sbuf + t * SLOT;
#if HAVE_ASYNC_LDS
    async_copy16(p, l);
    async_copy16(p + 16, l + 8);
#else
    *(v8h*)l = *(const v8h*)p;
    *(v8h*)(l + 8) = *(const v8h*)(p + 16);
#endif
  }
}

// Double-buffered k-loop: stage next B tile (async) and prefetch next A fragment
// while running WMMAs on the current step.  Requires (K/32) even.
template <int JT>
__device__ __forceinline__ void mma_loop(const _Float16* A, int lda, int row0,
                                         const _Float16* Bt, int ldb, int colb, int K,
                                         _Float16* sb0, _Float16* sb1, v8f* acc) {
  int lane = threadIdx.x & 31;
  const int S = K >> 5;
  stage_B<JT>(Bt, ldb, colb, 0, sb0);
  v16h a_cur = load_A(A, lda, row0, 0);
  async_fence();
  __syncthreads();
  for (int s = 0; s < S; ++s) {
    v16h a_nxt;
    if (s + 1 < S) {
      stage_B<JT>(Bt, ldb, colb, (s + 1) << 5, ((s + 1) & 1) ? sb1 : sb0);
      a_nxt = load_A(A, lda, row0, (s + 1) << 5);
    }
    const _Float16* bb = (s & 1) ? sb1 : sb0;
#pragma unroll
    for (int j = 0; j < JT; ++j) {
      v16h b = *(const v16h*)(bb + (j * 32 + lane) * SLOT);
      acc[j] = wmma32(a_cur, b, acc[j]);
    }
    if (s + 1 < S) {
      async_fence();
      __syncthreads();
      a_cur = a_nxt;
    }
  }
}

// ---------------------------------------------------------------------------
// conversion / transpose kernels
// ---------------------------------------------------------------------------
__global__ __launch_bounds__(256) void cvt_f16_kernel(const float* in, _Float16* out, size_t n) {
  size_t i = (size_t)blockIdx.x * 256 + threadIdx.x;
  if (i < n) out[i] = (_Float16)in[i];
}

__global__ __launch_bounds__(256) void transpose_cvt_kernel(const float* in, _Float16* out,
                                                            int K, int N, int Npad) {
  size_t idx = (size_t)blockIdx.x * 256 + threadIdx.x;
  if (idx >= (size_t)Npad * K) return;
  int n = (int)(idx / K);
  int k = (int)(idx % K);
  out[idx] = (n < N) ? (_Float16)in[(size_t)k * N + n] : (_Float16)0.0f;
}

// out[b][c][r] = in[b][r][c]   (generic f16 64x64 LDS tile transpose)
__global__ __launch_bounds__(256) void transpose_f16_kernel(
    const _Float16* in, size_t in_row, size_t in_batch,
    _Float16* out, size_t out_row, size_t out_batch) {
  __shared__ _Float16 tile[64 * 65];
  int b = blockIdx.z;
  int r0 = blockIdx.x * 64, c0 = blockIdx.y * 64;
  const _Float16* src = in + (size_t)b * in_batch;
  _Float16* dst = out + (size_t)b * out_batch;
  int t = threadIdx.x;
#pragma unroll
  for (int i = 0; i < 16; ++i) {
    int e = t + i * 256;
    int rl = e >> 6, cl = e & 63;
    tile[rl * 65 + cl] = src[(size_t)(r0 + rl) * in_row + c0 + cl];
  }
  __syncthreads();
#pragma unroll
  for (int i = 0; i < 16; ++i) {
    int e = t + i * 256;
    int cl = e >> 6, rl = e & 63;
    dst[(size_t)(c0 + cl) * out_row + r0 + rl] = tile[rl * 65 + cl];
  }
}

// ---------------------------------------------------------------------------
// generic WMMA GEMM: C = A * B (+bias). B given as B^T. Block tile 128 x (JT*16).
// ---------------------------------------------------------------------------
template <int JT>
__global__ __launch_bounds__(256) void gemm_kernel(const _Float16* A, int lda,
                                                   const _Float16* Bt, int ldb, int K,
                                                   const float* bias, float* Cf,
                                                   _Float16* Ch, int ldc) {
  __shared__ alignas(32) _Float16 sB[2][JT * 32 * SLOT];
  int wave = threadIdx.x >> 5, lane = threadIdx.x & 31;
  int row0 = blockIdx.y * 128 + wave * 16;
  int colb = blockIdx.x * (JT * 16);
  v8f acc[JT];
#pragma unroll
  for (int j = 0; j < JT; ++j)
#pragma unroll
    for (int i = 0; i < 8; ++i) acc[j][i] = 0.0f;

  mma_loop<JT>(A, lda, row0, Bt, ldb, colb, K, sB[0], sB[1], acc);

  int r0 = row0 + (lane >> 4) * 8;
#pragma unroll
  for (int j = 0; j < JT; ++j) {
    int col = colb + j * 16 + (lane & 15);
    float bv = bias ? bias[col] : 0.0f;
#pragma unroll
    for (int i = 0; i < 8; ++i) {
      float v = acc[j][i] + bv;
      if (Cf) Cf[(size_t)(r0 + i) * ldc + col] = v;
      else    Ch[(size_t)(r0 + i) * ldc + col] = (_Float16)v;
    }
  }
}

// GT[b][d][g] = sum_m VT[b][d][m] * gwT[b][g][m]   (staged WMMA, K = 2048)
__global__ __launch_bounds__(256) void gt_kernel(const _Float16* VT, const _Float16* gwT,
                                                 _Float16* GT) {
  __shared__ alignas(32) _Float16 sB[2][4 * 32 * SLOT];
  int wave = threadIdx.x >> 5, lane = threadIdx.x & 31;
  int z = blockIdx.z;
  int row0 = blockIdx.y * 128 + wave * 16;  // d
  const _Float16* A  = VT  + (size_t)z * DIM * N_SEQ;
  const _Float16* Bt = gwT + (size_t)z * GPAD * N_SEQ;
  v8f acc[4];
#pragma unroll
  for (int j = 0; j < 4; ++j)
#pragma unroll
    for (int i = 0; i < 8; ++i) acc[j][i] = 0.0f;

  mma_loop<4>(A, N_SEQ, row0, Bt, N_SEQ, 0, N_SEQ, sB[0], sB[1], acc);

  _Float16* out = GT + (size_t)z * DIM * GPAD;
  int r0 = row0 + (lane >> 4) * 8;
#pragma unroll
  for (int j = 0; j < 4; ++j) {
    int col = j * 16 + (lane & 15);
#pragma unroll
    for (int i = 0; i < 8; ++i)
      out[(size_t)(r0 + i) * GPAD + col] = (_Float16)acc[j][i];
  }
}

// ---------------------------------------------------------------------------
// gw = softmax(gelu(gp[:,0:49]))  (one wave per row, 64-wide padded output)
// ---------------------------------------------------------------------------
__device__ __forceinline__ float gelu_exact(float x) {
  return 0.5f * x * (1.0f + erff(x * 0.70710678118654752f));
}

__global__ __launch_bounds__(256) void gelu_softmax_kernel(const _Float16* gp, _Float16* gw) {
  int wave = threadIdx.x >> 5, lane = threadIdx.x & 31;
  int row = blockIdx.x * 8 + wave;
  const _Float16* in = gp + (size_t)row * GPAD;
  bool v0ok = lane < GNUM, v1ok = (lane + 32) < GNUM;
  float g0 = v0ok ? gelu_exact((float)in[lane]) : -1e30f;
  float g1 = v1ok ? gelu_exact((float)in[lane + 32]) : -1e30f;
  float mx = fmaxf(g0, g1);
#pragma unroll
  for (int off = 16; off; off >>= 1) mx = fmaxf(mx, __shfl_xor(mx, off, 32));
  float e0 = v0ok ? __expf(g0 - mx) : 0.0f;
  float e1 = v1ok ? __expf(g1 - mx) : 0.0f;
  float s = e0 + e1;
#pragma unroll
  for (int off = 16; off; off >>= 1) s += __shfl_xor(s, off, 32);
  float inv = 1.0f / s;
  _Float16* out = gw + (size_t)row * GPAD;
  out[lane]      = (_Float16)(e0 * inv);
  out[lane + 32] = (_Float16)(e1 * inv);
}

// ---------------------------------------------------------------------------
// scores: S[b][n][m] = f16( scale * (q . k) * (gw_n . gw_m) )
// ---------------------------------------------------------------------------
__global__ __launch_bounds__(256) void scores_kernel(const _Float16* qkvh,
                                                     const _Float16* gwh, _Float16* S) {
  __shared__ alignas(32) _Float16 sB[2][8 * 32 * SLOT];
  int wave = threadIdx.x >> 5, lane = threadIdx.x & 31;
  int z = blockIdx.z;
  int row0 = blockIdx.y * 128 + wave * 16;  // n
  int colb = blockIdx.x * 128;              // m
  const _Float16* Aq = qkvh + (size_t)z * N_SEQ * TD;  // q at col 0
  const _Float16* Bk = Aq + DIM;                       // k at col 768
  const _Float16* Gz = gwh + (size_t)z * N_SEQ * GPAD;
  v8f accs[8], accw[8];
#pragma unroll
  for (int j = 0; j < 8; ++j)
#pragma unroll
    for (int i = 0; i < 8; ++i) { accs[j][i] = 0.0f; accw[j][i] = 0.0f; }

  mma_loop<8>(Aq, TD, row0, Bk, TD, colb, DIM, sB[0], sB[1], accs);       // 24 steps
  mma_loop<8>(Gz, GPAD, row0, Gz, GPAD, colb, GPAD, sB[0], sB[1], accw);  // 2 steps

  const float scale = 0.03608439182435161f;  // 768^-0.5
  _Float16* out = S + (size_t)z * N_SEQ * N_SEQ;
  int r0 = row0 + (lane >> 4) * 8;
#pragma unroll
  for (int j = 0; j < 8; ++j) {
    int col = colb + j * 16 + (lane & 15);
#pragma unroll
    for (int i = 0; i < 8; ++i)
      out[(size_t)(r0 + i) * N_SEQ + col] = (_Float16)(accs[j][i] * scale * accw[j][i]);
  }
}

// in-place row softmax over m = 2048 (one 256-thread block per row)
__global__ __launch_bounds__(256) void row_softmax_kernel(_Float16* S) {
  __shared__ float redm[8];
  __shared__ float reds[8];
  _Float16* p = S + (size_t)blockIdx.x * N_SEQ;
  int t = threadIdx.x, lane = t & 31, wave = t >> 5;
  float v[8];
  float mx = -1e30f;
#pragma unroll
  for (int i = 0; i < 8; ++i) { v[i] = (float)p[t + i * 256]; mx = fmaxf(mx, v[i]); }
#pragma unroll
  for (int off = 16; off; off >>= 1) mx = fmaxf(mx, __shfl_xor(mx, off, 32));
  if (lane == 0) redm[wave] = mx;
  __syncthreads();
  float bm = redm[0];
#pragma unroll
  for (int i = 1; i < 8; ++i) bm = fmaxf(bm, redm[i]);
  float s = 0.0f;
#pragma unroll
  for (int i = 0; i < 8; ++i) { v[i] = __expf(v[i] - bm); s += v[i]; }
#pragma unroll
  for (int off = 16; off; off >>= 1) s += __shfl_xor(s, off, 32);
  if (lane == 0) reds[wave] = s;
  __syncthreads();
  float tot = 0.0f;
#pragma unroll
  for (int i = 0; i < 8; ++i) tot += reds[i];
  float inv = 1.0f / tot;
#pragma unroll
  for (int i = 0; i < 8; ++i) p[t + i * 256] = (_Float16)(v[i] * inv);
}

// Y = (1-a) * P@V + a * gw@G^T   (f16 out, [16384 x 768])
__global__ __launch_bounds__(256) void attn_out_kernel(const _Float16* S, const _Float16* VT,
                                                       const _Float16* gwh, const _Float16* GT,
                                                       const float* alpha, _Float16* Y) {
  __shared__ alignas(32) _Float16 sB[2][8 * 32 * SLOT];
  int wave = threadIdx.x >> 5, lane = threadIdx.x & 31;
  int z = blockIdx.z;
  int row0 = blockIdx.y * 128 + wave * 16;  // n
  int colb = blockIdx.x * 128;              // d
  const _Float16* P  = S   + (size_t)z * N_SEQ * N_SEQ;
  const _Float16* Vt = VT  + (size_t)z * DIM * N_SEQ;
  const _Float16* Gz = gwh + (size_t)z * N_SEQ * GPAD;
  const _Float16* Gt = GT  + (size_t)z * DIM * GPAD;
  v8f acc1[8], acc2[8];
#pragma unroll
  for (int j = 0; j < 8; ++j)
#pragma unroll
    for (int i = 0; i < 8; ++i) { acc1[j][i] = 0.0f; acc2[j][i] = 0.0f; }

  mma_loop<8>(P, N_SEQ, row0, Vt, N_SEQ, colb, N_SEQ, sB[0], sB[1], acc1);  // 64 steps
  mma_loop<8>(Gz, GPAD, row0, Gt, GPAD, colb, GPAD, sB[0], sB[1], acc2);    // 2 steps

  float aa = 1.0f / (1.0f + __expf(-alpha[0]));
  _Float16* out = Y + (size_t)z * N_SEQ * DIM;
  int r0 = row0 + (lane >> 4) * 8;
#pragma unroll
  for (int j = 0; j < 8; ++j) {
    int col = colb + j * 16 + (lane & 15);
#pragma unroll
    for (int i = 0; i < 8; ++i)
      out[(size_t)(r0 + i) * DIM + col] =
          (_Float16)((1.0f - aa) * acc1[j][i] + aa * acc2[j][i]);
  }
}

// ---------------------------------------------------------------------------
extern "C" void kernel_launch(void* const* d_in, const int* in_sizes, int n_in,
                              void* d_out, int out_size, void* d_ws, size_t ws_size,
                              hipStream_t stream) {
  (void)in_sizes; (void)n_in; (void)out_size; (void)ws_size;
  const float* x      = (const float*)d_in[0];
  const float* W_qkv  = (const float*)d_in[1];
  const float* b_qkv  = (const float*)d_in[2];
  const float* W_proj = (const float*)d_in[3];
  const float* b_proj = (const float*)d_in[4];
  const float* W_gp   = (const float*)d_in[5];
  const float* alpha  = (const float*)d_in[6];
  float* out = (float*)d_out;

  char* ws = (char*)d_ws;
  size_t off = 0;
  auto alloc = [&](size_t bytes) { char* p = ws + off; off += bytes; return p; };
  _Float16* Xh     = (_Float16*)alloc((size_t)MROWS * DIM * 2);    // also used as Y
  _Float16* WqkvT  = (_Float16*)alloc((size_t)TD * DIM * 2);
  _Float16* WprojT = (_Float16*)alloc((size_t)DIM * DIM * 2);
  _Float16* WgpT   = (_Float16*)alloc((size_t)128 * DIM * 2);      // padded 49 -> 128 rows
  _Float16* qkvh   = (_Float16*)alloc((size_t)MROWS * TD * 2);
  _Float16* gph    = (_Float16*)alloc((size_t)MROWS * GPAD * 2);
  _Float16* gwh    = (_Float16*)alloc((size_t)MROWS * GPAD * 2);
  _Float16* gwT    = (_Float16*)alloc((size_t)B_SZ * GPAD * N_SEQ * 2);
  _Float16* VT     = (_Float16*)alloc((size_t)B_SZ * DIM * N_SEQ * 2);
  _Float16* GT     = (_Float16*)alloc((size_t)B_SZ * DIM * GPAD * 2);
  _Float16* Sbuf   = (_Float16*)alloc((size_t)B_SZ * N_SEQ * N_SEQ * 2);
  _Float16* Y      = Xh;  // Xh dead after qkv GEMM

  // 1. f16 conversions (+ weight transposes, zero-padded W_gp)
  {
    size_t n = (size_t)MROWS * DIM;
    cvt_f16_kernel<<<(unsigned)((n + 255) / 256), 256, 0, stream>>>(x, Xh, n);
  }
  transpose_cvt_kernel<<<(TD * DIM + 255) / 256, 256, 0, stream>>>(W_qkv, WqkvT, DIM, TD, TD);
  transpose_cvt_kernel<<<(DIM * DIM + 255) / 256, 256, 0, stream>>>(W_proj, WprojT, DIM, DIM, DIM);
  transpose_cvt_kernel<<<(128 * DIM + 255) / 256, 256, 0, stream>>>(W_gp, WgpT, DIM, GNUM, 128);

  // 2. qkv = x @ W_qkv + b_qkv   (f16 out, [16384 x 2304])
  gemm_kernel<8><<<dim3(TD / 128, MROWS / 128, 1), 256, 0, stream>>>(
      Xh, DIM, WqkvT, DIM, DIM, b_qkv, nullptr, qkvh, TD);

  // 3. gp = v @ W_gp  (64-wide padded; only first 49 cols meaningful)
  gemm_kernel<4><<<dim3(1, MROWS / 128, 1), 256, 0, stream>>>(
      qkvh + 1536, TD, WgpT, DIM, DIM, nullptr, nullptr, gph, GPAD);

  // 4. gw = softmax(gelu(gp)) over 49 groups, zero-padded to 64
  gelu_softmax_kernel<<<MROWS / 8, 256, 0, stream>>>(gph, gwh);

  // 5. V^T [b][d][m] and gw^T [b][g][m]
  transpose_f16_kernel<<<dim3(N_SEQ / 64, DIM / 64, B_SZ), 256, 0, stream>>>(
      qkvh + 1536, TD, (size_t)N_SEQ * TD, VT, N_SEQ, (size_t)DIM * N_SEQ);
  transpose_f16_kernel<<<dim3(N_SEQ / 64, GPAD / 64, B_SZ), 256, 0, stream>>>(
      gwh, GPAD, (size_t)N_SEQ * GPAD, gwT, N_SEQ, (size_t)GPAD * N_SEQ);

  // 6. G^T[b][d][g] = sum_m V^T[d][m] gw[m][g]
  gt_kernel<<<dim3(1, DIM / 128, B_SZ), 256, 0, stream>>>(VT, gwT, GT);

  // 7. S = scale * (q k^T) .* (gw gw^T)   (f16, [b][2048][2048]; L2-resident)
  scores_kernel<<<dim3(N_SEQ / 128, N_SEQ / 128, B_SZ), 256, 0, stream>>>(qkvh, gwh, Sbuf);

  // 8. P = row-softmax(S) in place
  row_softmax_kernel<<<B_SZ * N_SEQ, 256, 0, stream>>>(Sbuf);

  // 9. Y = (1-a) P@V + a gw@G^T
  attn_out_kernel<<<dim3(DIM / 128, N_SEQ / 128, B_SZ), 256, 0, stream>>>(
      Sbuf, VT, gwh, GT, alpha, Y);

  // 10. out = Y @ W_proj + b_proj   (f32 -> d_out)
  gemm_kernel<8><<<dim3(DIM / 128, MROWS / 128, 1), 256, 0, stream>>>(
      Y, DIM, WprojT, DIM, DIM, b_proj, out, nullptr, DIM);
}
